// VisualAttention_5411658793545
// MI455X (gfx1250) — matrix operand order repaired and checked by reading, not compiled, not run on hardware
//
#include <hip/hip_runtime.h>
#include <hip/hip_bf16.h>

// ---------------- types ----------------
typedef __bf16 bf16_t;
typedef __attribute__((ext_vector_type(16))) __bf16 v16bf;
typedef __attribute__((ext_vector_type(8)))  float  v8f;
typedef __attribute__((ext_vector_type(4)))  unsigned int u32x4;
typedef __attribute__((ext_vector_type(4)))  int  i32x4;
typedef __attribute__((ext_vector_type(8)))  int  i32x8;

#define SQ    1024
#define BATCH 8
#define HEADS 16
#define HDIM  80
#define DPAD  96
#define EMB   1280
#define E3    3840
#define NB    (BATCH * HEADS)   // 128 attention batches

#if __has_builtin(__builtin_amdgcn_tensor_load_to_lds) && \
    __has_builtin(__builtin_amdgcn_s_wait_tensorcnt)
#define HAVE_TDM 1
#else
#define HAVE_TDM 0
#endif

__device__ inline v8f v8f_zero() {
    v8f z;
#pragma unroll
    for (int i = 0; i < 8; ++i) z[i] = 0.0f;
    return z;
}

union Frag16 {
    v16bf v;
    uint4 q[2];
};

// load 16 bf16 as two contiguous 8-element (16B) chunks
__device__ inline v16bf load_frag_2x8(const bf16_t* p0, const bf16_t* p1) {
    Frag16 f;
    f.q[0] = *(const uint4*)p0;
    f.q[1] = *(const uint4*)p1;
    return f.v;
}
// load 16 contiguous bf16 (32B)
__device__ inline v16bf load_frag_16(const bf16_t* p) {
    Frag16 f;
    f.q[0] = *(const uint4*)p;
    f.q[1] = *(const uint4*)(p + 8);
    return f.v;
}

__device__ inline v8f wmma_bf16(v16bf a, v16bf b, v8f c) {
    return __builtin_amdgcn_wmma_f32_16x16x32_bf16(
        /*neg_a=*/false, a, /*neg_b=*/false, b,
        /*c_mod=*/(short)0, c, /*reuse_a=*/false, /*reuse_b=*/false);
}

#if HAVE_TDM
// ---------------------------------------------------------------------
// Tensor Data Mover: 2D tile load (global -> LDS), data_size = 2 bytes.
// D# bit layout per CDNA5 ISA 8.3/8.4:
//   group0: [1:0]=count=1, [63:32]=lds_addr, [120:64]=global_addr,
//           [127:126]=type=2
//   group1: [17:16]=data_size(1 => 2B), [79:48]=tensor_dim0,
//           [111:80]=tensor_dim1, [127:112]=tile_dim0, [143:128]=tile_dim1,
//           [207:160]=tensor_dim0_stride
// ---------------------------------------------------------------------
__device__ inline void tdm_load_2d(unsigned lds_off, const void* gptr,
                                   unsigned tensor_d0, unsigned tensor_d1,
                                   unsigned stride0,
                                   unsigned tile_d0, unsigned tile_d1)
{
    unsigned long long ga = (unsigned long long)(size_t)gptr;
    u32x4 g0;
    g0[0] = 1u;                                            // count=1, user
    g0[1] = lds_off;                                       // lds_addr
    g0[2] = (unsigned)(ga & 0xFFFFFFFFu);                  // global_addr lo
    g0[3] = (unsigned)((ga >> 32) & 0x1FFFFFFu) | (2u << 30); // hi + type=2
    i32x8 g1;
    g1[0] = (int)(1u << 16);                               // data_size = 2B
    g1[1] = (int)((tensor_d0 & 0xFFFFu) << 16);
    g1[2] = (int)(((tensor_d0 >> 16) & 0xFFFFu) | ((tensor_d1 & 0xFFFFu) << 16));
    g1[3] = (int)(((tensor_d1 >> 16) & 0xFFFFu) | ((tile_d0 & 0xFFFFu) << 16));
    g1[4] = (int)(tile_d1 & 0xFFFFu);                      // tile_dim1, tile_dim2=0
    g1[5] = (int)stride0;                                  // dim0 stride lo
    g1[6] = 0;
    g1[7] = 0;
    i32x4 z4 = {0, 0, 0, 0};
#if defined(__clang_major__) && __clang_major__ >= 23
    i32x8 z8 = {0, 0, 0, 0, 0, 0, 0, 0};
    __builtin_amdgcn_tensor_load_to_lds(g0, g1, z4, z4, z8, 0);
#else
    __builtin_amdgcn_tensor_load_to_lds(g0, g1, z4, z4, 0);
#endif
}
__device__ inline unsigned lds_offset_of(const void* p) {
    // flat LDS address: its low 32 bits are the LDS offset (ISA 10.2 aperture rules)
    return (unsigned)(size_t)p;
}
#endif // HAVE_TDM

// =====================================================================
// Kernel 1: in_proj GEMM  C[8192 x 3840] = x @ w_in + b_in
// scatter epilogue -> q_ws [n][s][DPAD] (scaled), k_ws [n][s][DPAD],
//                     vt_ws [n][d][s]
// =====================================================================
__global__ __launch_bounds__(256) void in_proj_kernel(
    const float* __restrict__ x, const float* __restrict__ w_in,
    const float* __restrict__ b_in,
    bf16_t* __restrict__ q_ws, bf16_t* __restrict__ k_ws,
    bf16_t* __restrict__ vt_ws)
{
    __shared__ bf16_t Asm[128][32];   // A tile row-major [m][k]
    __shared__ bf16_t Bsm[128][32];   // B tile transposed [n][k]

    const int tid  = threadIdx.x;
    const int lane = tid & 31;
    const int wave = tid >> 5;
    const int bm   = blockIdx.x;          // 64 tiles of M
    const int bn   = blockIdx.y;          // 30 tiles of N
    const int wm   = wave & 3;            // 32-row group
    const int wn   = wave >> 2;           // 64-col group
    const int hf   = lane >> 4;           // lane half
    const int l16  = lane & 15;
    const int kb0  = hf * 8;

    v8f acc[2][4];
#pragma unroll
    for (int i = 0; i < 2; ++i)
#pragma unroll
        for (int j = 0; j < 4; ++j) acc[i][j] = v8f_zero();

    for (int k0 = 0; k0 < EMB; k0 += 32) {
        // ---- load phase: issue all global loads first (clause-friendly) ----
        float4 fa[4], fb[4];
#pragma unroll
        for (int i = 0; i < 4; ++i) {
            int idx = tid + i * 256;          // A: 0..1023 float4 slots
            int r   = idx >> 3;               // 8 float4 per row
            int c4  = (idx & 7) * 4;
            fa[i] = *(const float4*)(x + (size_t)(bm * 128 + r) * EMB + k0 + c4);
        }
#pragma unroll
        for (int i = 0; i < 4; ++i) {
            int idx = tid + i * 256;          // B: one float4 along j per thread
            int k   = idx & 31;
            int j0  = (idx >> 5) * 4;
            fb[i] = *(const float4*)(w_in + (size_t)(k0 + k) * E3 + bn * 128 + j0);
        }
        // branchless prefetch of next K-step (clamped to last tile)
        {
            int kp = (k0 + 32 < EMB) ? (k0 + 32) : k0;
            __builtin_prefetch(x + (size_t)(bm * 128 + (tid >> 1)) * EMB + kp + (tid & 1) * 16, 0, 1);
            __builtin_prefetch(w_in + (size_t)(kp + (tid >> 5)) * E3 + bn * 128 + lane * 4, 0, 1);
        }
        // ---- store phase: convert + LDS stores ----
#pragma unroll
        for (int i = 0; i < 4; ++i) {
            int idx = tid + i * 256;
            int r   = idx >> 3;
            int c4  = (idx & 7) * 4;
            Asm[r][c4 + 0] = (bf16_t)fa[i].x;
            Asm[r][c4 + 1] = (bf16_t)fa[i].y;
            Asm[r][c4 + 2] = (bf16_t)fa[i].z;
            Asm[r][c4 + 3] = (bf16_t)fa[i].w;
        }
#pragma unroll
        for (int i = 0; i < 4; ++i) {
            int idx = tid + i * 256;
            int k   = idx & 31;
            int j0  = (idx >> 5) * 4;
            Bsm[j0 + 0][k] = (bf16_t)fb[i].x;
            Bsm[j0 + 1][k] = (bf16_t)fb[i].y;
            Bsm[j0 + 2][k] = (bf16_t)fb[i].z;
            Bsm[j0 + 3][k] = (bf16_t)fb[i].w;
        }
        __syncthreads();

        v16bf bfr[4];
#pragma unroll
        for (int tc = 0; tc < 4; ++tc)
            bfr[tc] = load_frag_16(&Bsm[wn * 64 + tc * 16 + l16][hf * 16]);
#pragma unroll
        for (int tr = 0; tr < 2; ++tr) {
            const bf16_t* ap = &Asm[wm * 32 + tr * 16 + l16][0];
            v16bf afr = load_frag_16(ap + hf * 16);
#pragma unroll
            for (int tc = 0; tc < 4; ++tc)
                acc[tr][tc] = wmma_bf16(afr, bfr[tc], acc[tr][tc]);
        }
        __syncthreads();
    }

    // ---- scatter epilogue ----
    const float qscale = 0.11180339887498948f;   // 1/sqrt(80)
#pragma unroll
    for (int tr = 0; tr < 2; ++tr) {
#pragma unroll
        for (int tc = 0; tc < 4; ++tc) {
            int j    = bn * 128 + wn * 64 + tc * 16 + l16;  // < 3840
            float bias = b_in[j];
            int h = j / 240;
            int c = j % 240;
#pragma unroll
            for (int r = 0; r < 8; ++r) {
                int m  = bm * 128 + wm * 32 + tr * 16 + hf * 8 + r;
                int s  = m >> 3;        // seq index
                int bb = m & 7;         // batch index
                int n  = bb * HEADS + h;
                float v = acc[tr][tc][r] + bias;
                if (c < HDIM) {
                    q_ws[((size_t)n * SQ + s) * DPAD + c] = (bf16_t)(v * qscale);
                } else if (c < 2 * HDIM) {
                    k_ws[((size_t)n * SQ + s) * DPAD + (c - HDIM)] = (bf16_t)v;
                } else {
                    vt_ws[((size_t)n * HDIM + (c - 2 * HDIM)) * SQ + s] = (bf16_t)v;
                }
            }
        }
    }
}

// =====================================================================
// Kernel 2: flash attention.  block = (n, q-tile of 128 rows), 8 waves,
// each wave owns 16 q rows.  kv in tiles of 64, TDM double-buffered.
// =====================================================================
__global__ __launch_bounds__(256) void attn_kernel(
    const bf16_t* __restrict__ q_ws, const bf16_t* __restrict__ k_ws,
    const bf16_t* __restrict__ vt_ws, bf16_t* __restrict__ ctx_ws)
{
    __shared__ bf16_t Ksm[2][64][DPAD];   // 24 KB  [buf][kv][d]
    __shared__ bf16_t Vsm[2][HDIM][64];   // 20 KB  [buf][d][kv]
    __shared__ bf16_t Psm[8][16][64];     // 16 KB  per-wave P staging

    const int tid  = threadIdx.x;
    const int lane = tid & 31;
    const int wave = tid >> 5;
    const int n    = blockIdx.x;          // 0..127
    const int qt   = blockIdx.y;          // 0..7
    const int hf   = lane >> 4;
    const int l16  = lane & 15;
    const int kb0  = hf * 8;
    const int qrow0 = qt * 128 + wave * 16;
    const int NT   = SQ / 64;             // 16 kv tiles

    // preload Q fragments (A-matrix, 3 K-steps of 32 over DPAD=96)
    v16bf qf[3];
#pragma unroll
    for (int ks = 0; ks < 3; ++ks) {
        const bf16_t* qp = q_ws + ((size_t)n * SQ + qrow0 + l16) * DPAD + ks * 32;
        qf[ks] = load_frag_16(qp + hf * 16);
    }

    v8f ctx[5];
#pragma unroll
    for (int dc = 0; dc < 5; ++dc) ctx[dc] = v8f_zero();
    float mrow[8], lrow[8];
#pragma unroll
    for (int r = 0; r < 8; ++r) { mrow[r] = -1.0e30f; lrow[r] = 0.0f; }

#if HAVE_TDM
    // prologue: TDM-load tile 0 into buffer 0 (one wave drives the DMA)
    if (wave == 0) {
        tdm_load_2d(lds_offset_of(&Ksm[0][0][0]),
                    k_ws + ((size_t)n * SQ + 0) * DPAD,
                    DPAD, 64, DPAD, DPAD, 64);
        tdm_load_2d(lds_offset_of(&Vsm[0][0][0]),
                    vt_ws + (size_t)n * HDIM * SQ + 0,
                    64, HDIM, SQ, 64, HDIM);
    }
#endif

    for (int t = 0; t < NT; ++t) {
        const int buf = t & 1;
#if HAVE_TDM
        if (wave == 0) {
            if (t + 1 < NT) {   // kick off next tile into the other buffer
                const int nb = (t + 1) & 1;
                tdm_load_2d(lds_offset_of(&Ksm[nb][0][0]),
                            k_ws + ((size_t)n * SQ + (t + 1) * 64) * DPAD,
                            DPAD, 64, DPAD, DPAD, 64);
                tdm_load_2d(lds_offset_of(&Vsm[nb][0][0]),
                            vt_ws + (size_t)n * HDIM * SQ + (t + 1) * 64,
                            64, HDIM, SQ, 64, HDIM);
                __builtin_amdgcn_s_wait_tensorcnt((short)2); // tile t done
            } else {
                __builtin_amdgcn_s_wait_tensorcnt((short)0);
            }
        }
        __syncthreads();   // tile t visible to all waves
#else
        {
            const uint4* src = (const uint4*)(k_ws + ((size_t)n * SQ + t * 64) * DPAD);
            uint4* dst = (uint4*)&Ksm[buf][0][0];
            for (int i = tid; i < 64 * DPAD / 8; i += 256) dst[i] = src[i];
        }
        for (int i = tid; i < HDIM * 8; i += 256) {
            int d = i >> 3, c = (i & 7) * 8;
            *(uint4*)&Vsm[buf][d][c] =
                *(const uint4*)(vt_ws + ((size_t)n * HDIM + d) * SQ + t * 64 + c);
        }
        __syncthreads();
#endif

        // ---- S = Q K^T : 4 kv-subtiles x 3 K-steps ----
        v8f sa[4];
#pragma unroll
        for (int tc = 0; tc < 4; ++tc) sa[tc] = v8f_zero();
#pragma unroll
        for (int ks = 0; ks < 3; ++ks) {
#pragma unroll
            for (int tc = 0; tc < 4; ++tc) {
                v16bf bfrag = load_frag_16(&Ksm[buf][tc * 16 + l16][ks * 32 + hf * 16]);
                sa[tc] = wmma_bf16(qf[ks], bfrag, sa[tc]);
            }
        }

        // ---- online softmax (per q row, width-16 shuffle reductions) ----
        float scale_r[8];
#pragma unroll
        for (int r = 0; r < 8; ++r) {
            float mx = sa[0][r];
#pragma unroll
            for (int tc = 1; tc < 4; ++tc) mx = fmaxf(mx, sa[tc][r]);
#pragma unroll
            for (int off = 1; off < 16; off <<= 1)
                mx = fmaxf(mx, __shfl_xor(mx, off, 16));
            float mnew = fmaxf(mrow[r], mx);
            float sc   = __expf(mrow[r] - mnew);
            float rs   = 0.0f;
#pragma unroll
            for (int tc = 0; tc < 4; ++tc) {
                float p = __expf(sa[tc][r] - mnew);
                sa[tc][r] = p;
                rs += p;
            }
#pragma unroll
            for (int off = 1; off < 16; off <<= 1)
                rs += __shfl_xor(rs, off, 16);
            lrow[r]    = lrow[r] * sc + rs;
            mrow[r]    = mnew;
            scale_r[r] = sc;
        }
        // rescale running context
#pragma unroll
        for (int dc = 0; dc < 5; ++dc)
#pragma unroll
            for (int r = 0; r < 8; ++r) ctx[dc][r] *= scale_r[r];

        // ---- P (C-layout) -> per-wave LDS (A-layout friendly) ----
#pragma unroll
        for (int r = 0; r < 8; ++r)
#pragma unroll
            for (int tc = 0; tc < 4; ++tc)
                Psm[wave][hf * 8 + r][tc * 16 + l16] = (bf16_t)sa[tc][r];

        // ---- ctx += P @ V : 2 K-steps x 5 d-subtiles ----
#pragma unroll
        for (int ks2 = 0; ks2 < 2; ++ks2) {
            const bf16_t* pp = &Psm[wave][l16][ks2 * 32];
            v16bf af = load_frag_16(pp + hf * 16);
#pragma unroll
            for (int dc = 0; dc < 5; ++dc) {
                v16bf bfrag = load_frag_16(&Vsm[buf][dc * 16 + l16][ks2 * 32 + hf * 16]);
                ctx[dc] = wmma_bf16(af, bfrag, ctx[dc]);
            }
        }
        __syncthreads();   // all waves done with buffer t before it is refilled
    }

    // ---- finalize: ctx/l -> ctx_ws in [sq*b][E] layout ----
    const int bb = n >> 4;       // n = bb*16 + h
    const int h  = n & 15;
#pragma unroll
    for (int r = 0; r < 8; ++r) {
        int s = qrow0 + hf * 8 + r;
        float inv = 1.0f / lrow[r];
#pragma unroll
        for (int dc = 0; dc < 5; ++dc) {
            float v = ctx[dc][r] * inv;
            ctx_ws[((size_t)s * BATCH + bb) * EMB + h * HDIM + dc * 16 + l16] = (bf16_t)v;
        }
    }
}

// =====================================================================
// Kernel 3: out_proj GEMM  out[8192 x 1280] = ctx @ w_out + b_out (fp32)
// =====================================================================
__global__ __launch_bounds__(256) void out_proj_kernel(
    const bf16_t* __restrict__ ctx_ws, const float* __restrict__ w_out,
    const float* __restrict__ b_out, float* __restrict__ out)
{
    __shared__ bf16_t Asm[128][32];
    __shared__ bf16_t Bsm[128][32];   // transposed [n][k]

    const int tid  = threadIdx.x;
    const int lane = tid & 31;
    const int wave = tid >> 5;
    const int bm   = blockIdx.x;      // 64
    const int bn   = blockIdx.y;      // 10
    const int wm   = wave & 3;
    const int wn   = wave >> 2;
    const int hf   = lane >> 4;
    const int l16  = lane & 15;
    const int kb0  = hf * 8;

    v8f acc[2][4];
#pragma unroll
    for (int i = 0; i < 2; ++i)
#pragma unroll
        for (int j = 0; j < 4; ++j) acc[i][j] = v8f_zero();

    for (int k0 = 0; k0 < EMB; k0 += 32) {
        // ---- load phase ----
        uint4 qa[2];
        float4 fb[4];
#pragma unroll
        for (int i = 0; i < 2; ++i) {
            int idx = tid + i * 256;          // 0..511 uint4 slots (A, bf16)
            int r = idx >> 2, c = (idx & 3) * 8;
            qa[i] = *(const uint4*)(ctx_ws + (size_t)(bm * 128 + r) * EMB + k0 + c);
        }
#pragma unroll
        for (int i = 0; i < 4; ++i) {
            int idx = tid + i * 256;
            int k   = idx & 31;
            int j0  = (idx >> 5) * 4;
            fb[i] = *(const float4*)(w_out + (size_t)(k0 + k) * EMB + bn * 128 + j0);
        }
        {
            int kp = (k0 + 32 < EMB) ? (k0 + 32) : k0;
            __builtin_prefetch(ctx_ws + (size_t)(bm * 128 + (tid >> 1)) * EMB + kp + (tid & 1) * 16, 0, 1);
            __builtin_prefetch(w_out + (size_t)(kp + (tid >> 5)) * EMB + bn * 128 + lane * 4, 0, 1);
        }
        // ---- store phase ----
#pragma unroll
        for (int i = 0; i < 2; ++i) {
            int idx = tid + i * 256;
            int r = idx >> 2, c = (idx & 3) * 8;
            *(uint4*)&Asm[r][c] = qa[i];
        }
#pragma unroll
        for (int i = 0; i < 4; ++i) {
            int idx = tid + i * 256;
            int k   = idx & 31;
            int j0  = (idx >> 5) * 4;
            Bsm[j0 + 0][k] = (bf16_t)fb[i].x;
            Bsm[j0 + 1][k] = (bf16_t)fb[i].y;
            Bsm[j0 + 2][k] = (bf16_t)fb[i].z;
            Bsm[j0 + 3][k] = (bf16_t)fb[i].w;
        }
        __syncthreads();

        v16bf bfr[4];
#pragma unroll
        for (int tc = 0; tc < 4; ++tc)
            bfr[tc] = load_frag_16(&Bsm[wn * 64 + tc * 16 + l16][hf * 16]);
#pragma unroll
        for (int tr = 0; tr < 2; ++tr) {
            const bf16_t* ap = &Asm[wm * 32 + tr * 16 + l16][0];
            v16bf afr = load_frag_16(ap + hf * 16);
#pragma unroll
            for (int tc = 0; tc < 4; ++tc)
                acc[tr][tc] = wmma_bf16(afr, bfr[tc], acc[tr][tc]);
        }
        __syncthreads();
    }

#pragma unroll
    for (int tr = 0; tr < 2; ++tr) {
#pragma unroll
        for (int tc = 0; tc < 4; ++tc) {
            int j = bn * 128 + wn * 64 + tc * 16 + l16;
            float bias = b_out[j];
#pragma unroll
            for (int r = 0; r < 8; ++r) {
                int m = bm * 128 + wm * 32 + tr * 16 + hf * 8 + r;
                out[(size_t)m * EMB + j] = acc[tr][tc][r] + bias;
            }
        }
    }
}

// =====================================================================
// launch
// =====================================================================
extern "C" void kernel_launch(void* const* d_in, const int* in_sizes, int n_in,
                              void* d_out, int out_size, void* d_ws, size_t ws_size,
                              hipStream_t stream) {
    (void)in_sizes; (void)n_in; (void)out_size; (void)ws_size;
    const float* x     = (const float*)d_in[0];
    const float* w_in  = (const float*)d_in[1];
    const float* b_in  = (const float*)d_in[2];
    const float* w_out = (const float*)d_in[3];
    const float* b_out = (const float*)d_in[4];
    float* out = (float*)d_out;

    bf16_t* q_ws   = (bf16_t*)d_ws;                       // [128][1024][96]
    bf16_t* k_ws   = q_ws  + (size_t)NB * SQ * DPAD;      // [128][1024][96]
    bf16_t* vt_ws  = k_ws  + (size_t)NB * SQ * DPAD;      // [128][80][1024]
    bf16_t* ctx_ws = vt_ws + (size_t)NB * HDIM * SQ;      // [8192][1280]

    // zero q/k pad columns (graph-capture-legal memset node)
    hipMemsetAsync(q_ws, 0, (size_t)2 * NB * SQ * DPAD * sizeof(bf16_t), stream);

    in_proj_kernel<<<dim3(64, 30), 256, 0, stream>>>(x, w_in, b_in, q_ws, k_ws, vt_ws);
    attn_kernel<<<dim3(NB, SQ / 128), 256, 0, stream>>>(q_ws, k_ws, vt_ws, ctx_ws);
    out_proj_kernel<<<dim3(64, 10), 256, 0, stream>>>(ctx_ws, w_out, b_out, out);
}
